// NICDecoderModule_45002667328012
// MI455X (gfx1250) — matrix operand order repaired and checked by reading, not compiled, not run on hardware
//
#include <hip/hip_runtime.h>
#include <stdint.h>

// ---------- types ----------
typedef __bf16 bf16_t;
typedef bf16_t v16bf __attribute__((ext_vector_type(16)));
typedef float  v8f   __attribute__((ext_vector_type(8)));
typedef unsigned short u16;

// ---------- problem constants ----------
#define NN    128     // batch
#define PIX   196     // 14*14
#define ENC   2048
#define VV    10000
#define EE    512
#define DD    512
#define TT    31      // MAXLEN-1
#define KCAT  3072    // E + ENC + D  (concat input to the LSTM gate GEMM)
#define GDIM  2048    // 4*D

// ---------- helpers ----------
__device__ __forceinline__ u16 f2bf(float f) {
  union { float f; unsigned u; } v; v.f = f;
  unsigned u = v.u;
  u += 0x7FFFu + ((u >> 16) & 1u);   // round-to-nearest-even
  return (u16)(u >> 16);
}

__device__ __forceinline__ float sigmoidf(float x) {
  return 1.0f / (1.0f + __expf(-x));
}

union Frag32 { v16bf v; uint4 q[2]; };

__device__ __forceinline__ uint4 ld16(const u16* p) {
  return *reinterpret_cast<const uint4*>(p);
}

// One 16x16 output tile, A (MxK bf16, row-major K-contig), B = W rows (N x K,
// row-major K-contig, i.e. B-matrix of X@W^T). Caller pre-offsets:
//   A0 = A + (m0 + (lane&15))*lda + (lane>>4)*16
//   B0 = W + (n0 + (lane&15))*ldb + (lane>>4)*16
__device__ __forceinline__ v8f wmma_dot(const u16* __restrict__ A0,
                                        const u16* __restrict__ B0, int K) {
  v8f acc = {0.f,0.f,0.f,0.f,0.f,0.f,0.f,0.f};
  for (int k = 0; k < K; k += 32) {
    Frag32 a, b;
    a.q[0] = ld16(A0 + k);        // K = k + khalf*16 .. +7    (VGPR 0-3)
    a.q[1] = ld16(A0 + k + 8);    // K = k + khalf*16 + 8 ..   (VGPR 4-7)
    b.q[0] = ld16(B0 + k);        // K = k + khalf*16 .. +7
    b.q[1] = ld16(B0 + k + 8);    // K = k + khalf*16 + 8 ..
    acc = __builtin_amdgcn_wmma_f32_16x16x32_bf16(false, a.v, false, b.v,
                                                  (short)0, acc, false, false);
  }
  return acc;
}

// ---------- setup kernels ----------

// Stable argsort(-cl) via O(N^2) ranking (N=128, one block).
__global__ void prep_sort_kernel(const int* __restrict__ capt_len,
                                 int* __restrict__ sortind, int* __restrict__ declen,
                                 float* __restrict__ out_declen,
                                 float* __restrict__ out_sortind) {
  __shared__ int cl[NN];
  int i = threadIdx.x;
  cl[i] = capt_len[i];
  __syncthreads();
  int ci = cl[i];
  int r = 0;
  for (int j = 0; j < NN; ++j) {
    int cj = cl[j];
    if (cj > ci || (cj == ci && j < i)) ++r;
  }
  sortind[r] = i;
  declen[r]  = ci - 1;
  out_sortind[r] = (float)i;
  out_declen[r]  = (float)(ci - 1);
}

__global__ void gather_capt_kernel(const int* __restrict__ enc_capt,
                                   const int* __restrict__ sortind,
                                   int* __restrict__ captS,
                                   float* __restrict__ out_capt) {
  int idx = blockIdx.x * 256 + threadIdx.x;   // 128*32
  int m = idx >> 5, t = idx & 31;
  int v = enc_capt[sortind[m] * 32 + t];
  captS[idx] = v;
  out_capt[idx] = (float)v;
}

__global__ void f2bf_copy_kernel(const float* __restrict__ src,
                                 u16* __restrict__ dst, int n) {
  for (int i = blockIdx.x * blockDim.x + threadIdx.x; i < n;
       i += gridDim.x * blockDim.x)
    dst[i] = f2bf(src[i]);
}

// Wcat[n][k] : k<2560 -> W_ih[n][k], else W_hh[n][k-2560]   (n < 2048, k < 3072)
__global__ void build_wcat_kernel(const float* __restrict__ Wih,
                                  const float* __restrict__ Whh,
                                  u16* __restrict__ dst) {
  int idx = blockIdx.x * 256 + threadIdx.x;
  int n = idx / KCAT, k = idx - n * KCAT;
  float v = (k < EE + ENC) ? Wih[n * (EE + ENC) + k]
                           : Whh[n * DD + (k - EE - ENC)];
  dst[idx] = f2bf(v);
}

// mean over 196 pixels, sorted batch order; f32 + bf16 copies
__global__ void mean_kernel(const float* __restrict__ enc_feature,
                            const int* __restrict__ sortind,
                            float* __restrict__ meanf, u16* __restrict__ meanb) {
  int e = blockIdx.x * blockDim.x + threadIdx.x;   // 0..ENC-1
  int m = blockIdx.y;
  const float* src = enc_feature + (size_t)sortind[m] * (PIX * ENC) + e;
  float s = 0.f;
  #pragma unroll 4
  for (int p = 0; p < PIX; ++p) s += src[(size_t)p * ENC];
  s *= (1.0f / (float)PIX);
  meanf[m * ENC + e] = s;
  meanb[m * ENC + e] = f2bf(s);
}

// ---------- GEMM kernels (one wave = one 16x16 tile) ----------

// h0/c0 = mean_feat @ W_init{h,c}^T + b   (M=128, K=2048, N=512); y: 0=h 1=c
__global__ __launch_bounds__(256) void gemm_init_kernel(
    const u16* __restrict__ meanb, const u16* __restrict__ Wh,
    const u16* __restrict__ Wc, const float* __restrict__ bh,
    const float* __restrict__ bc, float* __restrict__ hf,
    float* __restrict__ cf, u16* __restrict__ hrec, u16* __restrict__ X) {
  int wave = threadIdx.x >> 5, lane = threadIdx.x & 31;
  int tile = blockIdx.x * 8 + wave;
  const int TN = DD / 16;
  int tm = tile / TN, tn = tile % TN;
  int khalf = lane >> 4, ln = lane & 15;
  bool do_c = (blockIdx.y == 1);
  const u16* W = do_c ? Wc : Wh;
  const u16* A0 = meanb + (tm * 16 + ln) * ENC + khalf * 16;
  const u16* B0 = W + (tn * 16 + ln) * ENC + khalf * 16;
  v8f acc = wmma_dot(A0, B0, ENC);
  const float* bias = do_c ? bc : bh;
  #pragma unroll
  for (int j = 0; j < 8; ++j) {
    int m = tm * 16 + j + khalf * 8;
    int n = tn * 16 + ln;
    float v = acc[j] + bias[n];
    if (do_c) {
      cf[m * DD + n] = v;
    } else {
      hf[m * DD + n] = v;
      u16 hb = f2bf(v);
      hrec[m * DD + n] = hb;
      X[m * KCAT + EE + ENC + n] = hb;
    }
  }
}

// awe = sigmoid(h @ W_fbeta^T + b) * mean_feat   (M=128, K=512, N=2048)
__global__ __launch_bounds__(256) void gemm_gate_kernel(
    const u16* __restrict__ hrec, const u16* __restrict__ Wfb,
    const float* __restrict__ bfb, const float* __restrict__ meanf,
    u16* __restrict__ X) {
  int wave = threadIdx.x >> 5, lane = threadIdx.x & 31;
  int tile = blockIdx.x * 8 + wave;
  const int TN = ENC / 16;
  int tm = tile / TN, tn = tile % TN;
  int khalf = lane >> 4, ln = lane & 15;
  const u16* A0 = hrec + (tm * 16 + ln) * DD + khalf * 16;
  const u16* B0 = Wfb + (tn * 16 + ln) * DD + khalf * 16;
  v8f acc = wmma_dot(A0, B0, DD);
  #pragma unroll
  for (int j = 0; j < 8; ++j) {
    int m = tm * 16 + j + khalf * 8;
    int n = tn * 16 + ln;
    float gate = sigmoidf(acc[j] + bfb[n]);
    X[m * KCAT + EE + n] = f2bf(gate * meanf[m * ENC + n]);
  }
}

// token embedding gather into X[:, 0:512]
__global__ void embed_gather_kernel(const float* __restrict__ emb,
                                    const int* __restrict__ captS, int t,
                                    u16* __restrict__ X) {
  int idx = blockIdx.x * 256 + threadIdx.x;   // 128*512
  int m = idx >> 9, e = idx & 511;
  int tok = captS[m * 32 + t];
  X[m * KCAT + e] = f2bf(emb[tok * EE + e]);
}

// g = [e_t|awe|h] @ [W_ih|W_hh]^T + b_ih + b_hh  (M=128, K=3072, N=2048)
__global__ __launch_bounds__(256) void gemm_g_kernel(
    const u16* __restrict__ X, const u16* __restrict__ Wcat,
    const float* __restrict__ bih, const float* __restrict__ bhh,
    float* __restrict__ gbuf) {
  int wave = threadIdx.x >> 5, lane = threadIdx.x & 31;
  int tile = blockIdx.x * 8 + wave;
  const int TN = GDIM / 16;
  int tm = tile / TN, tn = tile % TN;
  int khalf = lane >> 4, ln = lane & 15;
  const u16* A0 = X + (tm * 16 + ln) * KCAT + khalf * 16;
  const u16* B0 = Wcat + (tn * 16 + ln) * KCAT + khalf * 16;
  v8f acc = wmma_dot(A0, B0, KCAT);
  #pragma unroll
  for (int j = 0; j < 8; ++j) {
    int m = tm * 16 + j + khalf * 8;
    int n = tn * 16 + ln;
    gbuf[m * GDIM + n] = acc[j] + bih[n] + bhh[n];
  }
}

// LSTM pointwise update + masking
__global__ void lstm_update_kernel(const float* __restrict__ gbuf,
                                   const int* __restrict__ declen, int t,
                                   float* __restrict__ hf, float* __restrict__ cf,
                                   u16* __restrict__ hrec, u16* __restrict__ hnew,
                                   u16* __restrict__ X) {
  int idx = blockIdx.x * blockDim.x + threadIdx.x;   // 128*512
  int m = idx >> 9, d = idx & 511;
  float gi = gbuf[m * GDIM + d];
  float gf = gbuf[m * GDIM + 512 + d];
  float gg = gbuf[m * GDIM + 1024 + d];
  float go = gbuf[m * GDIM + 1536 + d];
  float c_old = cf[idx], h_old = hf[idx];
  float c_new = sigmoidf(gf) * c_old + sigmoidf(gi) * tanhf(gg);
  float h_new = sigmoidf(go) * tanhf(c_new);
  bool msk = t < declen[m];
  float h2 = msk ? h_new : h_old;
  float c2 = msk ? c_new : c_old;
  hf[idx] = h2;
  cf[idx] = c2;
  u16 hb = f2bf(h2);
  hrec[idx] = hb;
  X[m * KCAT + EE + ENC + d] = hb;
  hnew[idx] = f2bf(h_new);   // preds use unmasked h_new
}

// preds = mask * (h_new @ W_fc^T + b_fc)   (M=128, K=512, N=10000)
__global__ __launch_bounds__(256) void gemm_preds_kernel(
    const u16* __restrict__ hnew, const u16* __restrict__ Wfc,
    const float* __restrict__ bfc, const int* __restrict__ declen, int t,
    float* __restrict__ pred) {
  int wave = threadIdx.x >> 5, lane = threadIdx.x & 31;
  int tile = blockIdx.x * 8 + wave;
  const int TN = VV / 16;   // 625
  int tm = tile / TN, tn = tile % TN;
  int khalf = lane >> 4, ln = lane & 15;
  const u16* A0 = hnew + (tm * 16 + ln) * DD + khalf * 16;
  const u16* B0 = Wfc + (tn * 16 + ln) * DD + khalf * 16;
  v8f acc = wmma_dot(A0, B0, DD);
  #pragma unroll
  for (int j = 0; j < 8; ++j) {
    int m = tm * 16 + j + khalf * 8;
    int n = tn * 16 + ln;
    bool msk = t < declen[m];
    pred[((size_t)m * TT + t) * VV + n] = msk ? (acc[j] + bfc[n]) : 0.f;
  }
}

// alphas = mask / P  broadcast over P
__global__ void alphas_kernel(const int* __restrict__ declen,
                              float* __restrict__ out) {
  int idx = blockIdx.x * 256 + threadIdx.x;   // 128*31*196
  int m = idx / (TT * PIX);
  int r = idx - m * (TT * PIX);
  int t = r / PIX;
  out[idx] = (t < declen[m]) ? (1.0f / (float)PIX) : 0.f;
}

// ---------- host ----------
extern "C" void kernel_launch(void* const* d_in, const int* in_sizes, int n_in,
                              void* d_out, int out_size, void* d_ws, size_t ws_size,
                              hipStream_t stream) {
  (void)in_sizes; (void)n_in; (void)out_size; (void)ws_size;

  const float* enc_feature = (const float*)d_in[0];
  const int*   enc_capt    = (const int*)  d_in[1];
  const int*   capt_len    = (const int*)  d_in[2];
  const float* emb         = (const float*)d_in[3];
  const float* W_ih        = (const float*)d_in[4];
  const float* b_ih        = (const float*)d_in[5];
  const float* W_hh        = (const float*)d_in[6];
  const float* b_hh        = (const float*)d_in[7];
  const float* W_inith     = (const float*)d_in[8];
  const float* b_inith     = (const float*)d_in[9];
  const float* W_initc     = (const float*)d_in[10];
  const float* b_initc     = (const float*)d_in[11];
  const float* W_fbeta     = (const float*)d_in[12];
  const float* b_fbeta     = (const float*)d_in[13];
  const float* W_fc        = (const float*)d_in[14];
  const float* b_fc        = (const float*)d_in[15];

  // workspace carve (~33 MB)
  char* p = (char*)d_ws;
  auto carve = [&](size_t bytes) -> char* {
    char* r = p;
    p += (bytes + 255) & ~(size_t)255;
    return r;
  };
  int*   sortind = (int*)  carve((size_t)NN * 4);
  int*   declen  = (int*)  carve((size_t)NN * 4);
  int*   captS   = (int*)  carve((size_t)NN * 32 * 4);
  float* meanf   = (float*)carve((size_t)NN * ENC * 4);
  u16*   meanb   = (u16*)  carve((size_t)NN * ENC * 2);
  u16*   wcat    = (u16*)  carve((size_t)GDIM * KCAT * 2);
  u16*   wfb     = (u16*)  carve((size_t)ENC * DD * 2);
  u16*   wih0    = (u16*)  carve((size_t)DD * ENC * 2);
  u16*   wic0    = (u16*)  carve((size_t)DD * ENC * 2);
  u16*   wfc     = (u16*)  carve((size_t)VV * DD * 2);
  u16*   X       = (u16*)  carve((size_t)NN * KCAT * 2);
  u16*   hrec    = (u16*)  carve((size_t)NN * DD * 2);
  u16*   hnew    = (u16*)  carve((size_t)NN * DD * 2);
  float* hf      = (float*)carve((size_t)NN * DD * 4);
  float* cf      = (float*)carve((size_t)NN * DD * 4);
  float* gbuf    = (float*)carve((size_t)NN * GDIM * 4);

  // output layout (flat f32, tuple order)
  float* out          = (float*)d_out;
  float* out_pred     = out;                                  // 128*31*10000
  float* out_captf    = out_pred + (size_t)NN * TT * VV;      // 128*32
  float* out_declen   = out_captf + (size_t)NN * 32;          // 128
  float* out_alphas   = out_declen + NN;                      // 128*31*196
  float* out_sortind  = out_alphas + (size_t)NN * TT * PIX;   // 128

  // setup
  prep_sort_kernel<<<1, 128, 0, stream>>>(capt_len, sortind, declen,
                                          out_declen, out_sortind);
  gather_capt_kernel<<<16, 256, 0, stream>>>(enc_capt, sortind, captS, out_captf);
  f2bf_copy_kernel<<<1024, 256, 0, stream>>>(W_fbeta, wfb, ENC * DD);
  f2bf_copy_kernel<<<1024, 256, 0, stream>>>(W_inith, wih0, DD * ENC);
  f2bf_copy_kernel<<<1024, 256, 0, stream>>>(W_initc, wic0, DD * ENC);
  f2bf_copy_kernel<<<2048, 256, 0, stream>>>(W_fc, wfc, VV * DD);
  build_wcat_kernel<<<(GDIM * KCAT) / 256, 256, 0, stream>>>(W_ih, W_hh, wcat);
  mean_kernel<<<dim3(ENC / 256, NN), 256, 0, stream>>>(enc_feature, sortind,
                                                       meanf, meanb);
  gemm_init_kernel<<<dim3(32, 2), 256, 0, stream>>>(meanb, wih0, wic0, b_inith,
                                                    b_initc, hf, cf, hrec, X);

  // recurrent steps
  for (int t = 0; t < TT; ++t) {
    embed_gather_kernel<<<256, 256, 0, stream>>>(emb, captS, t, X);
    gemm_gate_kernel<<<128, 256, 0, stream>>>(hrec, wfb, b_fbeta, meanf, X);
    gemm_g_kernel<<<128, 256, 0, stream>>>(X, wcat, b_ih, b_hh, gbuf);
    lstm_update_kernel<<<256, 256, 0, stream>>>(gbuf, declen, t, hf, cf,
                                                hrec, hnew, X);
    gemm_preds_kernel<<<625, 256, 0, stream>>>(hnew, wfc, b_fc, declen, t,
                                               out_pred);
  }

  alphas_kernel<<<(NN * TT * PIX) / 256, 256, 0, stream>>>(declen, out_alphas);
}
